// DARNN_Encoder_46351287058676
// MI455X (gfx1250) — matrix operand order repaired and checked by reading, not compiled, not run on hardware
//
#include <hip/hip_runtime.h>

// ---------------------------------------------------------------------------
// DA-RNN encoder for MI455X (gfx1250, wave32, WMMA).
// B=2048, T=100, I=1, H=512.  Persistent kernel: 1 workgroup = 16 batch rows,
// iterates all 100 timesteps with LDS-resident state.
// ---------------------------------------------------------------------------

typedef __attribute__((ext_vector_type(16))) __bf16       v16bf;
typedef __attribute__((ext_vector_type(8)))  float        v8f;
typedef __attribute__((ext_vector_type(4)))  unsigned int v4u;
typedef __attribute__((ext_vector_type(8)))  unsigned int v8u;

#define T_LEN 100
#define H_DIM 512
#define G_DIM 2048          // 4*H
#define KB    16            // H_DIM / 32 k-blocks per WMMA chain
#define ROWS  16            // batch rows per workgroup (one WMMA M-tile)
#define OUT_T 16            // last 16 timesteps are the output
#define NTHREADS 256        // 8 waves of 32
#define LOG2E 1.4426950408889634f

// ---- fast math helpers (v_exp_f32 / v_rcp_f32) ----------------------------
__device__ __forceinline__ float fast_exp(float x) {
  return __builtin_amdgcn_exp2f(x * LOG2E);
}
__device__ __forceinline__ float fast_rcp(float x) {
  return __builtin_amdgcn_rcpf(x);
}
__device__ __forceinline__ float fast_sigm(float x) {
  return fast_rcp(1.0f + fast_exp(-x));
}
__device__ __forceinline__ float fast_tanh(float x) {
  // tanh(x) = 1 - 2/(e^{2x}+1)
  return 1.0f - 2.0f * fast_rcp(fast_exp(2.0f * x) + 1.0f);
}

// f32 -> bf16 bits, round-to-nearest-even
__device__ __forceinline__ unsigned short f2bf(float f) {
  unsigned int u = __builtin_bit_cast(unsigned int, f);
  unsigned int r = u + 0x7FFFu + ((u >> 16) & 1u);
  return (unsigned short)(r >> 16);
}

// ---- WMMA fragment helpers ------------------------------------------------
// A operand: 16x32 bf16 tile of h held row-major in LDS (row stride H_DIM).
// Per ISA layout: lanes 0-15 = rows, K {0..7,16..23}; lanes 16-31: K {8..15,24..31}.
__device__ __forceinline__ v16bf load_a_frag(const unsigned short* hbf, int kb) {
  int lane = threadIdx.x & 31;
  int half = lane >> 4;
  int m    = lane & 15;
  const unsigned short* base = hbf + m * H_DIM + kb * 32 + half * 8;
  v4u lo = *(const v4u*)(base);        // ds_load_b128
  v4u hi = *(const v4u*)(base + 16);   // ds_load_b128
  v8u u = { lo.x, lo.y, lo.z, lo.w, hi.x, hi.y, hi.z, hi.w };
  return __builtin_bit_cast(v16bf, u);
}

// B operand: packed layout wp[ntile][kb][n(16)][k(32)] bf16, K contiguous.
// Lanes 0-15: column n, K {0..7,16..23}; lanes 16-31: column n, K {8..15,24..31}.
__device__ __forceinline__ v16bf load_b_frag(const unsigned short* wp, int ntile, int kb) {
  int lane = threadIdx.x & 31;
  int half = lane >> 4;
  int n    = lane & 15;
  const unsigned short* base =
      wp + ((((long)ntile * KB + kb) * 16 + n) << 5) + half * 8;
  v4u lo = *(const v4u*)(base);        // global_load_b128
  v4u hi = *(const v4u*)(base + 16);   // global_load_b128
  v8u u = { lo.x, lo.y, lo.z, lo.w, hi.x, hi.y, hi.z, hi.w };
  return __builtin_bit_cast(v16bf, u);
}

__device__ __forceinline__ v8f wmma_bf16(v16bf a, v16bf b, v8f c) {
  return __builtin_amdgcn_wmma_f32_16x16x32_bf16(
      /*neg_a=*/false, a, /*neg_b=*/false, b,
      /*c_mod=*/(short)0, c, /*reuse_a=*/false, /*reuse_b=*/false);
}

// ---------------------------------------------------------------------------
// Pre-pass: pack W2 (H x H) and W_hh (4H x H) f32 row-major into bf16
// B-operand tiles: wp[ntile][kb][n][k] = W[ntile*16+n][kb*32+k].
// (gemm is y = h @ W^T, so B[k][ncol] = W[ncol][k]; K is contiguous in W rows)
// ---------------------------------------------------------------------------
__global__ void pack_weights_kernel(const float* __restrict__ W2,
                                    const float* __restrict__ Whh,
                                    unsigned short* __restrict__ w2p,
                                    unsigned short* __restrict__ whp) {
  long i = (long)blockIdx.x * blockDim.x + threadIdx.x;
  const long n2 = (long)H_DIM * H_DIM;
  const long nh = (long)G_DIM * H_DIM;
  if (i < n2) {
    long kk = i & 31;
    long rest = i >> 5;
    long n  = rest & 15; rest >>= 4;
    long kb = rest % KB;
    long nt = rest / KB;
    w2p[i] = f2bf(W2[(nt * 16 + n) * H_DIM + kb * 32 + kk]);
  } else if (i < n2 + nh) {
    long j = i - n2;
    long kk = j & 31;
    long rest = j >> 5;
    long n  = rest & 15; rest >>= 4;
    long kb = rest % KB;
    long nt = rest / KB;
    whp[j] = f2bf(Whh[(nt * 16 + n) * H_DIM + kb * 32 + kk]);
  }
}

// ---------------------------------------------------------------------------
// Main persistent kernel: one workgroup per 16 batch rows.
// ---------------------------------------------------------------------------
__global__ __launch_bounds__(NTHREADS, 1)
void darnn_encoder_kernel(const float* __restrict__ x,     // (B,T,1)
                          const float* __restrict__ W1w,   // (H,1)
                          const float* __restrict__ W1b,   // (H)
                          const float* __restrict__ W2b,   // (H)
                          const float* __restrict__ vw,    // (1,H)
                          const float* __restrict__ Wih,   // (4H,1)
                          const float* __restrict__ bih,   // (4H)
                          const float* __restrict__ bhh,   // (4H)
                          const unsigned short* __restrict__ w2p,
                          const unsigned short* __restrict__ whp,
                          float* __restrict__ out)         // (B,16,H)
{
  // LDS-resident per-workgroup state (~136 KB of 320 KB/WGP)
  __shared__ __align__(16) unsigned short h_bf[2][ROWS * H_DIM]; // 2x16 KB, double buffer
  __shared__ __align__(16) float c_st[ROWS * H_DIM];             // 32 KB
  __shared__ __align__(16) float q_st[ROWS * H_DIM];             // 32 KB
  __shared__ float x_st[ROWS * T_LEN];                           // 6.4 KB
  __shared__ float logit[ROWS * T_LEN];                          // 6.4 KB
  __shared__ float w1s[H_DIM], w1bs[H_DIM], vws[H_DIM], w2bs[H_DIM]; // 8 KB
  __shared__ float gbase[G_DIM], wihs[G_DIM];                    // 16 KB
  __shared__ float wmean[ROWS];

  const int tid  = threadIdx.x;
  const int wave = tid >> 5;
  const int lane = tid & 31;
  const int halfl = lane >> 4;
  const int ncol  = lane & 15;
  const int b0 = blockIdx.x * ROWS;

  // ---- one-time loads / init ----
  for (int i = tid; i < H_DIM; i += NTHREADS) {
    w1s[i]  = W1w[i];
    w1bs[i] = W1b[i];
    vws[i]  = vw[i];
    w2bs[i] = W2b[i];
  }
  for (int i = tid; i < G_DIM; i += NTHREADS) {
    gbase[i] = bih[i] + bhh[i];
    wihs[i]  = Wih[i];
  }
  for (int i = tid; i < ROWS * T_LEN; i += NTHREADS) {
    int m = i / T_LEN, s = i - m * T_LEN;
    x_st[i] = x[(long)(b0 + m) * T_LEN + s];
  }
  for (int i = tid; i < ROWS * H_DIM; i += NTHREADS) {
    c_st[i] = 0.0f;
    h_bf[0][i] = 0;
    h_bf[1][i] = 0;
  }
  __syncthreads();

  int buf = 0;
  for (int t = 0; t < T_LEN; ++t) {
    const unsigned short* hcur = h_bf[buf];

    // ---------- Phase A: q = h @ W2^T + b2  (WMMA, 4 N-tiles per wave) ----
    {
      v8f a0 = {}, a1 = {}, a2 = {}, a3 = {};
      for (int kb = 0; kb < KB; ++kb) {
        v16bf afr = load_a_frag(hcur, kb);
        a0 = wmma_bf16(afr, load_b_frag(w2p, wave,      kb), a0);
        a1 = wmma_bf16(afr, load_b_frag(w2p, wave + 8,  kb), a1);
        a2 = wmma_bf16(afr, load_b_frag(w2p, wave + 16, kb), a2);
        a3 = wmma_bf16(afr, load_b_frag(w2p, wave + 24, kb), a3);
      }
      // D layout: VGPR r -> row r + 8*halfl, column ntile*16 + ncol
      #pragma unroll
      for (int r = 0; r < 8; ++r) {
        int m = r + 8 * halfl;
        q_st[m * H_DIM + (wave      ) * 16 + ncol] = a0[r] + w2bs[(wave      ) * 16 + ncol];
        q_st[m * H_DIM + (wave +  8 ) * 16 + ncol] = a1[r] + w2bs[(wave +  8 ) * 16 + ncol];
        q_st[m * H_DIM + (wave + 16 ) * 16 + ncol] = a2[r] + w2bs[(wave + 16 ) * 16 + ncol];
        q_st[m * H_DIM + (wave + 24 ) * 16 + ncol] = a3[r] + w2bs[(wave + 24 ) * 16 + ncol];
      }
    }
    __syncthreads();

    // ---------- Phase B: attention logits over suffix window --------------
    const int L = T_LEN - t;
    {
      int npair = ROWS * L;
      for (int p = tid; p < npair; p += NTHREADS) {
        int m = p & 15;
        int s = t + (p >> 4);
        float xv = x_st[m * T_LEN + s];
        const float* qrow = &q_st[m * H_DIM];
        float sum = 0.0f;
        #pragma unroll 4
        for (int h = 0; h < H_DIM; ++h) {
          float e = fast_tanh(fmaf(xv, w1s[h], w1bs[h]) + qrow[h]);
          sum = fmaf(vws[h], e, sum);
        }
        logit[m * T_LEN + s] = sum;   // +v_b omitted: softmax shift-invariant
      }
    }
    __syncthreads();

    // ---------- Phase B2: softmax over window -> weighted mean scalar -----
    if (tid < ROWS) {
      int m = tid;
      float mx = -1e30f;
      for (int s = t; s < T_LEN; ++s) mx = fmaxf(mx, logit[m * T_LEN + s]);
      float den = 0.0f, num = 0.0f;
      for (int s = t; s < T_LEN; ++s) {
        float e = fast_exp(logit[m * T_LEN + s] - mx);
        den += e;
        num = fmaf(e, x_st[m * T_LEN + s], num);
      }
      wmean[m] = num * fast_rcp(den * (float)L);
    }
    __syncthreads();

    // ---------- Phase C: gates = h @ Whh^T + wmean*Wih + bias; LSTM update
    {
      unsigned short* hnext = h_bf[buf ^ 1];
      for (int gsel = 0; gsel < 4; ++gsel) {
        int g = wave + gsel * 8;          // i-gate N-tile index, 0..31
        v8f ai = {}, af = {}, ag = {}, ao = {};
        for (int kb = 0; kb < KB; ++kb) {
          // prefetch next k-block of the i-gate weight tile (global_prefetch_b8)
          if (kb + 1 < KB)
            __builtin_prefetch(whp + (((long)g * KB + kb + 1) * 16) * 32, 0, 0);
          v16bf afr = load_a_frag(hcur, kb);
          ai = wmma_bf16(afr, load_b_frag(whp, g,      kb), ai);
          af = wmma_bf16(afr, load_b_frag(whp, g + 32, kb), af);
          ag = wmma_bf16(afr, load_b_frag(whp, g + 64, kb), ag);
          ao = wmma_bf16(afr, load_b_frag(whp, g + 96, kb), ao);
        }
        int jj = g * 16 + ncol;           // hidden-unit index 0..511
        #pragma unroll
        for (int r = 0; r < 8; ++r) {
          int m = r + 8 * halfl;
          float wm = wmean[m];
          float gi = ai[r] + gbase[jj       ] + wm * wihs[jj       ];
          float gf = af[r] + gbase[jj +  512] + wm * wihs[jj +  512];
          float gg = ag[r] + gbase[jj + 1024] + wm * wihs[jj + 1024];
          float go = ao[r] + gbase[jj + 1536] + wm * wihs[jj + 1536];
          float cOld = c_st[m * H_DIM + jj];
          float cNew = fast_sigm(gf) * cOld + fast_sigm(gi) * fast_tanh(gg);
          float hN   = fast_sigm(go) * fast_tanh(cNew);
          c_st[m * H_DIM + jj] = cNew;
          hnext[m * H_DIM + jj] = f2bf(hN);
          if (t >= T_LEN - OUT_T) {
            out[((long)(b0 + m) * OUT_T + (t - (T_LEN - OUT_T))) * H_DIM + jj] = hN;
          }
        }
      }
    }
    __syncthreads();
    buf ^= 1;
  }
}

// ---------------------------------------------------------------------------
// Inputs (setup_inputs order):
//  0 x(B,T,1) 1 W1_w(H,1) 2 W1_b(H) 3 W2_w(H,H) 4 W2_b(H) 5 v_w(1,H) 6 v_b(1)
//  7 W_ih(4H,1) 8 b_ih(4H) 9 W_hh(4H,H) 10 b_hh(4H)
// ---------------------------------------------------------------------------
extern "C" void kernel_launch(void* const* d_in, const int* in_sizes, int n_in,
                              void* d_out, int out_size, void* d_ws, size_t ws_size,
                              hipStream_t stream) {
  const float* x    = (const float*)d_in[0];
  const float* W1w  = (const float*)d_in[1];
  const float* W1b  = (const float*)d_in[2];
  const float* W2w  = (const float*)d_in[3];
  const float* W2b  = (const float*)d_in[4];
  const float* vw   = (const float*)d_in[5];
  const float* Wih  = (const float*)d_in[7];
  const float* bih  = (const float*)d_in[8];
  const float* Whh  = (const float*)d_in[9];
  const float* bhh  = (const float*)d_in[10];
  float* out = (float*)d_out;

  // workspace: packed bf16 weights (W2: 512 KB, W_hh: 2 MB)
  unsigned short* w2p = (unsigned short*)d_ws;
  unsigned short* whp = w2p + (long)H_DIM * H_DIM;

  const long npack = (long)H_DIM * H_DIM + (long)G_DIM * H_DIM;
  pack_weights_kernel<<<(int)((npack + NTHREADS - 1) / NTHREADS), NTHREADS, 0, stream>>>(
      W2w, Whh, w2p, whp);

  const int nbatch = in_sizes[0] / T_LEN;   // B = 2048
  darnn_encoder_kernel<<<nbatch / ROWS, NTHREADS, 0, stream>>>(
      x, W1w, W1b, W2b, vw, Wih, bih, bhh, w2p, whp, out);
}
